// DecoderOnlyTransformer_48584670052769
// MI455X (gfx1250) — matrix operand order repaired and checked by reading, not compiled, not run on hardware
//
#include <hip/hip_runtime.h>

typedef __bf16 bf16;
typedef bf16  v16bf __attribute__((ext_vector_type(16)));
typedef float v8f   __attribute__((ext_vector_type(8)));

#define TB   8
#define TS   2048
#define TD   256
#define THH  8
#define TDK  32
#define TF   1024
#define TV   512
#define TL   4
#define MTOK (TB*TS)

// Two 16-byte LDS chunks -> one 32-byte WMMA fragment register block.
__device__ __forceinline__ v16bf ld2x16B(const bf16* p0, const bf16* p1) {
    union { v16bf v; uint4 u[2]; } t;
    t.u[0] = *(const uint4*)p0;
    t.u[1] = *(const uint4*)p1;
    return t.v;
}

__device__ __forceinline__ v8f wmma_bf16(v16bf a, v16bf b, v8f c) {
    return __builtin_amdgcn_wmma_f32_16x16x32_bf16(false, a, false, b, (short)0, c, false, false);
}

// Load 8 f32 (2x global_load_b128), convert to bf16, one ds_store_b128.
__device__ __forceinline__ void cvt8_store(bf16* dst, const float* src) {
    float4 f0 = *(const float4*)src;
    float4 f1 = *(const float4*)(src + 4);
    union { uint4 u; bf16 h[8]; } pk;
    pk.h[0]=(bf16)f0.x; pk.h[1]=(bf16)f0.y; pk.h[2]=(bf16)f0.z; pk.h[3]=(bf16)f0.w;
    pk.h[4]=(bf16)f1.x; pk.h[5]=(bf16)f1.y; pk.h[6]=(bf16)f1.z; pk.h[7]=(bf16)f1.w;
    *(uint4*)dst = pk.u;
}

__device__ __forceinline__ unsigned pack2(float a, float b) {
    union { unsigned u; bf16 h[2]; } p;
    p.h[0] = (bf16)a; p.h[1] = (bf16)b;
    return p.u;
}

// ---------------------------------------------------------------------------
// Embedding + sinusoidal positional encoding
// ---------------------------------------------------------------------------
__launch_bounds__(256)
__global__ void embed_k(const int* __restrict__ tok, const float* __restrict__ emb,
                        float* __restrict__ x) {
    int row = blockIdx.x, t = threadIdx.x;
    int token = tok[row];
    int s = row & (TS - 1);
    int i2 = t >> 1;
    float div = __expf((float)(2 * i2) * (-9.210340371976184f / (float)TD));
    float ang = (float)s * div;
    float pe = (t & 1) ? __cosf(ang) : __sinf(ang);
    x[(size_t)row * TD + t] = emb[(size_t)token * TD + t] + pe;
}

// ---------------------------------------------------------------------------
// LayerNorm: one block per row of D=256
// ---------------------------------------------------------------------------
__launch_bounds__(256)
__global__ void ln_k(const float* __restrict__ in, const float* __restrict__ g,
                     const float* __restrict__ bb, float* __restrict__ out) {
    __shared__ float red[256];
    int row = blockIdx.x, t = threadIdx.x;
    float v = in[(size_t)row * TD + t];
    red[t] = v; __syncthreads();
    #pragma unroll
    for (int s = 128; s > 0; s >>= 1) { if (t < s) red[t] += red[t + s]; __syncthreads(); }
    float mu = red[0] * (1.f / TD);
    __syncthreads();
    float d = v - mu;
    red[t] = d * d; __syncthreads();
    #pragma unroll
    for (int s = 128; s > 0; s >>= 1) { if (t < s) red[t] += red[t + s]; __syncthreads(); }
    float var = red[0] * (1.f / TD);
    out[(size_t)row * TD + t] = d * rsqrtf(var + 1e-5f) * g[t] + bb[t];
}

// ---------------------------------------------------------------------------
// Tiled GEMM: C[M,N] = act(A[M,K] @ W[K,N] + bias + resid)
// BM=BN=64, BK=32, 256 threads = 8 waves, each wave -> 16x32 output strip.
// fp32 in memory, bf16 in LDS, f32 accumulate via v_wmma_f32_16x16x32_bf16.
// ---------------------------------------------------------------------------
template<bool RELU>
__launch_bounds__(256)
__global__ void gemm_k(const float* __restrict__ A, const float* __restrict__ W,
                       const float* __restrict__ bias, const float* __restrict__ resid,
                       float* __restrict__ C, int M, int N, int K) {
    __shared__ __align__(16) bf16 As[64][40];   // [m][k]
    __shared__ __align__(16) bf16 Bs[64][40];   // [n][k]  (W transposed)
    const int t    = threadIdx.x;
    const int lane = t & 31, w = t >> 5;
    const int hi   = lane >> 4, l16 = lane & 15;
    const int wm   = w & 3, wn = w >> 2;
    const int m0   = blockIdx.y * 64, n0 = blockIdx.x * 64;

    v8f acc0 = {0.f,0.f,0.f,0.f,0.f,0.f,0.f,0.f};
    v8f acc1 = {0.f,0.f,0.f,0.f,0.f,0.f,0.f,0.f};

    const int arow = t >> 2, akp = (t & 3) * 8;   // A tile: 64 rows x 32 k, 8 elems/thread
    const int bk2  = (t >> 4) * 2;                // W tile: k-row pairs 0,2,..,30
    const int bnp  = (t & 15) * 4;                // 4 n-columns per thread

    for (int kt = 0; kt < K; kt += 32) {
        // stage A tile (row-major): one ds_store_b128 per thread
        cvt8_store(&As[arow][akp], A + (size_t)(m0 + arow) * K + kt + akp);
        {   // stage W tile transposed -> Bs[n][k]; pack (k,k+1) pairs -> ds_store_b32
            const float* wp0 = W + (size_t)(kt + bk2) * N + n0 + bnp;
            const float* wp1 = wp0 + N;
            float4 f0 = *(const float4*)wp0;
            float4 f1 = *(const float4*)wp1;
            *(unsigned*)&Bs[bnp+0][bk2] = pack2(f0.x, f1.x);
            *(unsigned*)&Bs[bnp+1][bk2] = pack2(f0.y, f1.y);
            *(unsigned*)&Bs[bnp+2][bk2] = pack2(f0.z, f1.z);
            *(unsigned*)&Bs[bnp+3][bk2] = pack2(f0.w, f1.w);
        }
        __syncthreads();
        // A frag: lane<16 holds K0..15 ; lane>=16 holds K16..31
        v16bf a  = ld2x16B(&As[16*wm + l16][hi*16], &As[16*wm + l16][hi*16 + 8]);
        // B frag: lane<16 holds K0..15 ; lane>=16 holds K16..31 (per column n)
        v16bf b0 = ld2x16B(&Bs[32*wn + l16][hi*16],      &Bs[32*wn + l16][hi*16 + 8]);
        v16bf b1 = ld2x16B(&Bs[32*wn + 16 + l16][hi*16], &Bs[32*wn + 16 + l16][hi*16 + 8]);
        acc0 = wmma_bf16(a, b0, acc0);
        acc1 = wmma_bf16(a, b1, acc1);
        __syncthreads();
    }

    #pragma unroll
    for (int s = 0; s < 2; ++s) {
        v8f acc = s ? acc1 : acc0;
        int n = n0 + 32*wn + 16*s + l16;
        float bv = bias ? bias[n] : 0.f;
        #pragma unroll
        for (int r = 0; r < 8; ++r) {
            int m = m0 + 16*wm + r + 8*hi;      // C layout: lane-half selects M+8
            float val = acc[r] + bv;
            if (resid) val += resid[(size_t)m * N + n];
            if (RELU)  val = fmaxf(val, 0.f);
            C[(size_t)m * N + n] = val;
        }
    }
}

// ---------------------------------------------------------------------------
// Flash-style causal attention. One block per (64-query tile, head, batch).
// 128 threads = 4 waves; wave w owns query rows 16w..16w+15.
// Q,K,V are [MTOK, D] with head h at columns h*32..h*32+31.
// ---------------------------------------------------------------------------
__launch_bounds__(128)
__global__ void attn_k(const float* __restrict__ Qg, const float* __restrict__ Kg,
                       const float* __restrict__ Vg, float* __restrict__ O) {
    __shared__ __align__(16) bf16 Qs[64][40];   // [q][dk]
    __shared__ __align__(16) bf16 Ks[64][40];   // [key][dk]  == B frag layout for Q@K^T
    __shared__ __align__(16) bf16 Vt[32][72];   // [dk][key]  == B frag layout for P@V
    __shared__ __align__(16) bf16 Ps[64][72];   // [q][key]   == A frag source for P@V

    const int qt = blockIdx.x, h = blockIdx.y, b = blockIdx.z;
    const int t = threadIdx.x;
    const int lane = t & 31, w = t >> 5;
    const int hi = lane >> 4, l16 = lane & 15;

    const int srow = t >> 1, skp = (t & 1) * 16;   // row-major staging split
    const int vk2  = (t & 31) * 2;                 // key pair for V^T staging
    const int vdk  = (t >> 5) * 8;                 // 8 dk values per thread

    {   // stage Q tile once: two ds_store_b128 per thread
        const float* qp = Qg + (size_t)(b*TS + qt*64 + srow) * TD + h*TDK + skp;
        cvt8_store(&Qs[srow][skp],     qp);
        cvt8_store(&Qs[srow][skp + 8], qp + 8);
    }
    __syncthreads();
    v16bf aq = ld2x16B(&Qs[16*w + l16][hi*16], &Qs[16*w + l16][hi*16 + 8]);

    v8f o0 = {0.f,0.f,0.f,0.f,0.f,0.f,0.f,0.f};
    v8f o1 = {0.f,0.f,0.f,0.f,0.f,0.f,0.f,0.f};
    float m_i[8], l_i[8];
    #pragma unroll
    for (int r = 0; r < 8; ++r) { m_i[r] = -1e30f; l_i[r] = 0.f; }

    const float sc = 0.17677669529663687f;   // 1/sqrt(DK)

    for (int kt = 0; kt <= qt; ++kt) {
        {   // stage K row-major: two ds_store_b128 per thread
            const float* kpp = Kg + (size_t)(b*TS + kt*64 + srow) * TD + h*TDK + skp;
            cvt8_store(&Ks[srow][skp],     kpp);
            cvt8_store(&Ks[srow][skp + 8], kpp + 8);
        }
        {   // stage V transposed: pack key pairs -> 8x ds_store_b32 per thread
            const float* v0 = Vg + (size_t)(b*TS + kt*64 + vk2) * TD + h*TDK + vdk;
            const float* v1 = v0 + TD;
            float4 a0 = *(const float4*)v0, a1 = *(const float4*)(v0 + 4);
            float4 b0 = *(const float4*)v1, b1 = *(const float4*)(v1 + 4);
            float r0[8] = {a0.x,a0.y,a0.z,a0.w,a1.x,a1.y,a1.z,a1.w};
            float r1[8] = {b0.x,b0.y,b0.z,b0.w,b1.x,b1.y,b1.z,b1.w};
            #pragma unroll
            for (int j = 0; j < 8; ++j)
                *(unsigned*)&Vt[vdk + j][vk2] = pack2(r0[j], r1[j]);
        }
        __syncthreads();

        // scores: 16 q-rows x 64 keys = 4 WMMAs (K = DK = 32)
        v8f s_[4];
        #pragma unroll
        for (int sub = 0; sub < 4; ++sub) {
            v16bf bk = ld2x16B(&Ks[16*sub + l16][hi*16], &Ks[16*sub + l16][hi*16 + 8]);
            v8f z = {0.f,0.f,0.f,0.f,0.f,0.f,0.f,0.f};
            s_[sub] = wmma_bf16(aq, bk, z);
        }

        const bool diag = (kt == qt);
        #pragma unroll
        for (int r = 0; r < 8; ++r) {
            int qrow = qt*64 + 16*w + r + 8*hi;
            float e[4];
            #pragma unroll
            for (int sub = 0; sub < 4; ++sub) {
                float v = s_[sub][r] * sc;
                int key = kt*64 + sub*16 + l16;
                if (diag && key > qrow) v = -1e30f;
                e[sub] = v;
            }
            float tmax = fmaxf(fmaxf(e[0], e[1]), fmaxf(e[2], e[3]));
            #pragma unroll
            for (int off = 1; off < 16; off <<= 1)
                tmax = fmaxf(tmax, __shfl_xor(tmax, off, 32));
            float mnew = fmaxf(m_i[r], tmax);
            float corr = __expf(m_i[r] - mnew);
            m_i[r] = mnew;
            float rs = 0.f;
            int qr = 16*w + r + 8*hi;
            #pragma unroll
            for (int sub = 0; sub < 4; ++sub) {
                float p = __expf(e[sub] - mnew);
                rs += p;
                Ps[qr][16*sub + l16] = (bf16)p;
            }
            #pragma unroll
            for (int off = 1; off < 16; off <<= 1)
                rs += __shfl_xor(rs, off, 32);
            l_i[r] = l_i[r] * corr + rs;
            o0[r] *= corr;
            o1[r] *= corr;
        }

        // P @ V : two K=32 slices, two dk-subtiles (same-wave LDS RAW on Ps)
        #pragma unroll
        for (int ks = 0; ks < 2; ++ks) {
            v16bf ap  = ld2x16B(&Ps[16*w + l16][ks*32 + hi*16],  &Ps[16*w + l16][ks*32 + hi*16 + 8]);
            v16bf bv0 = ld2x16B(&Vt[l16][ks*32 + hi*16],        &Vt[l16][ks*32 + hi*16 + 8]);
            v16bf bv1 = ld2x16B(&Vt[16 + l16][ks*32 + hi*16],   &Vt[16 + l16][ks*32 + hi*16 + 8]);
            o0 = wmma_bf16(ap, bv0, o0);
            o1 = wmma_bf16(ap, bv1, o1);
        }
        __syncthreads();
    }

    #pragma unroll
    for (int r = 0; r < 8; ++r) {
        float inv = 1.f / l_i[r];
        int row = b*TS + qt*64 + 16*w + r + 8*hi;
        float* op = O + (size_t)row * TD + h*TDK;
        op[l16]      = o0[r] * inv;
        op[16 + l16] = o1[r] * inv;
    }
}

// ---------------------------------------------------------------------------
// Host orchestration
// ---------------------------------------------------------------------------
extern "C" void kernel_launch(void* const* d_in, const int* in_sizes, int n_in,
                              void* d_out, int out_size, void* d_ws, size_t ws_size,
                              hipStream_t stream) {
    (void)in_sizes; (void)n_in; (void)out_size; (void)ws_size;

    const int*   tokens = (const int*)d_in[0];
    const float* emb    = (const float*)d_in[1];
    // per-layer params: wq,bq,wk,bk,wv,bv,wo,bo,w1,b1,w2,b2,ln1_g,ln1_b,ln2_g,ln2_b
    auto LW = [&](int layer, int idx) { return (const float*)d_in[2 + layer * 16 + idx]; };
    const float* lnf_g   = (const float*)d_in[2 + TL * 16 + 0];
    const float* lnf_b   = (const float*)d_in[2 + TL * 16 + 1];
    const float* lm_head = (const float*)d_in[2 + TL * 16 + 2];

    const size_t NB = (size_t)MTOK * TD;       // 16384*256
    float* ws  = (float*)d_ws;
    float* x   = ws;
    float* t0  = ws + NB;
    float* q   = ws + 2 * NB;
    float* k   = ws + 3 * NB;
    float* v   = ws + 4 * NB;
    float* c   = ws + 5 * NB;
    float* ffh = q;                            // [MTOK, F] aliases q..c (exactly 4*NB)

    const dim3 blk256(256), blk128(128);
    const dim3 gD (TD  / 64, MTOK / 64);       // N=256
    const dim3 gF (TF  / 64, MTOK / 64);       // N=1024
    const dim3 gV (TV  / 64, MTOK / 64);       // N=512
    const dim3 gAt(TS / 64, THH, TB);          // (qtile, head, batch)

    embed_k<<<MTOK, blk256, 0, stream>>>(tokens, emb, x);

    for (int l = 0; l < TL; ++l) {
        gemm_k<false><<<gD, blk256, 0, stream>>>(x, LW(l,0), LW(l,1), nullptr, q, MTOK, TD, TD);
        gemm_k<false><<<gD, blk256, 0, stream>>>(x, LW(l,2), LW(l,3), nullptr, k, MTOK, TD, TD);
        gemm_k<false><<<gD, blk256, 0, stream>>>(x, LW(l,4), LW(l,5), nullptr, v, MTOK, TD, TD);

        attn_k<<<gAt, blk128, 0, stream>>>(q, k, v, c);

        gemm_k<false><<<gD, blk256, 0, stream>>>(c, LW(l,6), LW(l,7), x, t0, MTOK, TD, TD);
        ln_k<<<MTOK, blk256, 0, stream>>>(t0, LW(l,12), LW(l,13), x);

        gemm_k<true ><<<gF, blk256, 0, stream>>>(x, LW(l,8), LW(l,9), nullptr, ffh, MTOK, TF, TD);
        gemm_k<false><<<gD, blk256, 0, stream>>>(ffh, LW(l,10), LW(l,11), x, t0, MTOK, TD, TF);
        ln_k<<<MTOK, blk256, 0, stream>>>(t0, LW(l,14), LW(l,15), x);
    }

    ln_k<<<MTOK, blk256, 0, stream>>>(x, lnf_g, lnf_b, t0);
    gemm_k<false><<<gV, blk256, 0, stream>>>(t0, lm_head, nullptr, nullptr,
                                             (float*)d_out, MTOK, TV, TD);
}
